// DeformConv2d_v4_953482739891
// MI455X (gfx1250) — compile-verified, matrix operand order checked by reading, not run on hardware
//
#include <hip/hip_runtime.h>

// ---------------------------------------------------------------------------
// DCNv4 for MI455X (gfx1250, wave32, WMMA, TDM).
// Pipeline: pack(bf16) -> WMMA GEMM (xp|om fused, TDM-staged A) ->
//           deformable bilinear sample -> WMMA GEMM (output proj, bohw).
// ---------------------------------------------------------------------------

typedef __attribute__((ext_vector_type(16))) __bf16        v16bf;
typedef __attribute__((ext_vector_type(8)))  float         v8f;
typedef __attribute__((ext_vector_type(4)))  unsigned int  v4u;
typedef __attribute__((ext_vector_type(8)))  int           v8i;
typedef __attribute__((ext_vector_type(4)))  int           v4i;

#define BATCH   8
#define C_IN    256
#define H_DIM   48
#define W_DIM   48
#define HW      (H_DIM * W_DIM)          // 2304
#define M_TOTAL (BATCH * HW)             // 18432
#define OMC     432
#define N1      (C_IN + OMC)             // 688 (xp | om fused GEMM width)
#define GRP     16
#define GC      (C_IN / GRP)             // 16
#define NT1     (N1 / 16)                // 43 n-tiles, fused GEMM
#define NT2     (C_IN / 16)              // 16 n-tiles, output GEMM
#define MROWS   64                       // M rows per block (4 x 16 per wave)

__device__ __forceinline__ unsigned short f32_to_bf16(float f) {
    union { float f; unsigned int u; } c; c.f = f;
    unsigned int r = c.u + 0x7FFFu + ((c.u >> 16) & 1u);   // RNE
    return (unsigned short)(r >> 16);
}

// ---------------------------------------------------------------------------
// Tensor Data Mover: 1-D contiguous DMA global -> LDS (nelem bf16 elements).
// D# built per CDNA5 ISA 8.3/8.4: group0 {count=1, lds_addr, global_addr,
// type=2}; group1 {data_size=1(2B), tensor_dim0=tile_dim0=nelem}.
// ---------------------------------------------------------------------------
__device__ __forceinline__ void tdm_load_1d(const void* lds_dst,
                                            const void* gsrc,
                                            unsigned nelem)
{
    const unsigned lds_off = (unsigned)(uintptr_t)lds_dst;   // LDS aperture: [31:0]
    const unsigned long long ga = (unsigned long long)(uintptr_t)gsrc;
    v4u g0;
    g0[0] = 1u;                                   // count=1, user descriptor
    g0[1] = lds_off;                              // bits  63:32 lds_addr
    g0[2] = (unsigned)(ga & 0xFFFFFFFFu);         // bits  95:64 global_addr lo
    g0[3] = (unsigned)((ga >> 32) & 0x01FFFFFFu)  // bits 120:96 global_addr hi
          | (2u << 30);                           // bits 127:126 type=2 (image)
    v8i g1;
    g1[0] = (int)(1u << 16);                      // data_size=1 (2 bytes/elem)
    g1[1] = (int)((nelem & 0xFFFFu) << 16);       // tensor_dim0[15:0]  @ bits 63:48
    g1[2] = (int)((nelem >> 16) & 0xFFFFu);       // tensor_dim0[31:16] @ bits 79:64
    g1[3] = (int)((nelem & 0xFFFFu) << 16);       // tile_dim0          @ bits 127:112
    g1[4] = 0; g1[5] = 0; g1[6] = 0; g1[7] = 0;   // tile_dim1/2=0, strides=0
    v4i gz; gz[0] = 0; gz[1] = 0; gz[2] = 0; gz[3] = 0;
#if __clang_major__ >= 23
    v8i gz8; gz8[0]=0; gz8[1]=0; gz8[2]=0; gz8[3]=0; gz8[4]=0; gz8[5]=0; gz8[6]=0; gz8[7]=0;
    __builtin_amdgcn_tensor_load_to_lds(g0, g1, gz, gz, gz8, 0);
#else
    __builtin_amdgcn_tensor_load_to_lds(g0, g1, gz, gz, 0);
#endif
}

// ---------------------------------------------------------------------------
// x [B,C,HW] f32  ->  A [M = B*HW, K = C] bf16 (row-major), LDS-tiled transpose
// ---------------------------------------------------------------------------
__global__ void __launch_bounds__(256)
transpose_pack(const float* __restrict__ x, unsigned short* __restrict__ A)
{
    __shared__ float tile[32][33];
    const int b   = blockIdx.z;
    const int hw0 = blockIdx.x * 32;
    const int c0  = blockIdx.y * 32;
    const float* xb = x + (size_t)b * C_IN * HW;
    #pragma unroll
    for (int i = threadIdx.y; i < 32; i += 8)
        tile[i][threadIdx.x] = xb[(size_t)(c0 + i) * HW + hw0 + threadIdx.x];
    __syncthreads();
    #pragma unroll
    for (int i = threadIdx.y; i < 32; i += 8)
        A[(size_t)(b * HW + hw0 + i) * C_IN + c0 + threadIdx.x] =
            f32_to_bf16(tile[threadIdx.x][i]);
}

// ---------------------------------------------------------------------------
// Pack weights fragment-contiguous for WMMA-B: Bf[ntile][kstep][lane][8] dwords.
// Lane's whole 8-dword B fragment is 32 contiguous bytes -> 2 b128 loads.
// Dword j of lane L in k-step ks holds bf16 pair B[k0,n],B[k0+1,n] with
// n = nt*16 + (L&15), k0 = ks*32 + (j<4?0:16) + 2*(j&3) + 8*(L>>4)   (ISA 7.12.2)
// ---------------------------------------------------------------------------
__global__ void __launch_bounds__(256)
pack_weights(const float* __restrict__ w_value, const float* __restrict__ b_value,
             const float* __restrict__ w_om,    const float* __restrict__ b_om,
             const float* __restrict__ w_out,
             unsigned* __restrict__ Bf1, unsigned* __restrict__ Bf2,
             float* __restrict__ bias1)
{
    int idx = blockIdx.x * 256 + threadIdx.x;
    if (idx < NT1 * 2048) {
        const int j    = idx & 7;
        const int lane = (idx >> 3) & 31;
        const int ks   = (idx >> 8) & 7;
        const int nt   = idx >> 11;
        const int n    = nt * 16 + (lane & 15);
        const int k0   = ks * 32 + ((j < 4) ? 0 : 16) + (j & 3) * 2 + (lane >> 4) * 8;
        float lo, hi;
        if (n < C_IN) { lo = w_value[(size_t)n * C_IN + k0];
                        hi = w_value[(size_t)n * C_IN + k0 + 1]; }
        else          { lo = w_om[(size_t)(n - C_IN) * C_IN + k0];
                        hi = w_om[(size_t)(n - C_IN) * C_IN + k0 + 1]; }
        Bf1[idx] = (unsigned)f32_to_bf16(lo) | ((unsigned)f32_to_bf16(hi) << 16);
        return;
    }
    idx -= NT1 * 2048;
    if (idx < NT2 * 2048) {
        const int j    = idx & 7;
        const int lane = (idx >> 3) & 31;
        const int ks   = (idx >> 8) & 7;
        const int nt   = idx >> 11;
        const int n    = nt * 16 + (lane & 15);
        const int k0   = ks * 32 + ((j < 4) ? 0 : 16) + (j & 3) * 2 + (lane >> 4) * 8;
        const float lo = w_out[(size_t)n * C_IN + k0];
        const float hi = w_out[(size_t)n * C_IN + k0 + 1];
        Bf2[idx] = (unsigned)f32_to_bf16(lo) | ((unsigned)f32_to_bf16(hi) << 16);
        return;
    }
    idx -= NT2 * 2048;
    if (idx < N1)
        bias1[idx] = (idx < C_IN) ? b_value[idx] : b_om[idx - C_IN];
}

// ---------------------------------------------------------------------------
// WMMA core: one wave computes 64(M) x 16(N) in 4 accumulators.
// Per K-step: one fragment-packed B load (2 x b128), 4 A loads from LDS
// (2 x ds_load_b128 each), 4 x v_wmma_f32_16x16x32_bf16.
// ---------------------------------------------------------------------------
__device__ __forceinline__ void
wmma_64x16(const unsigned short* __restrict__ lds_a,
           const unsigned* __restrict__ Bf_nt, int lane, v8f acc[4])
{
    const int m   = lane & 15;
    const int khs = (lane >> 4) * 8;      // K-half offset in bf16 elements
    #pragma unroll
    for (int ks = 0; ks < 8; ++ks) {
        union { uint4 q[2]; v16bf v; } B;
        const uint4* bsrc = (const uint4*)(Bf_nt + ((size_t)ks * 32 + lane) * 8);
        B.q[0] = bsrc[0];
        B.q[1] = bsrc[1];
        #pragma unroll
        for (int ms = 0; ms < 4; ++ms) {
            union { uint4 q[2]; v16bf v; } A;
            const unsigned short* arow = lds_a + (ms * 16 + m) * C_IN + ks * 32 + khs;
            A.q[0] = *(const uint4*)(arow);
            A.q[1] = *(const uint4*)(arow + 16);
            acc[ms] = __builtin_amdgcn_wmma_f32_16x16x32_bf16(
                false, A.v, false, B.v, (short)0, acc[ms], false, false);
        }
    }
}

__device__ __forceinline__ void
stage_a_tdm(const unsigned short* __restrict__ A, int mt,
            unsigned short* __restrict__ lds_a, int wave)
{
    if (wave == 0) {
        tdm_load_1d(lds_a, A + (size_t)mt * MROWS * C_IN, MROWS * C_IN); // 32 KB
        __builtin_amdgcn_s_wait_tensorcnt(0);
    }
    __syncthreads();
}

// GEMM 1: xp|om fused.  D row-major [M][N1] f32 (+bias).
__global__ void __launch_bounds__(256)
gemm_xp_om(const unsigned short* __restrict__ A,
           const unsigned* __restrict__ Bf1,
           const float* __restrict__ bias1,
           float* __restrict__ xp_om)
{
    __shared__ unsigned short lds_a[MROWS * C_IN];     // 32 KB
    const int tid = threadIdx.x, mt = blockIdx.x;
    const int wave = tid >> 5, lane = tid & 31;
    stage_a_tdm(A, mt, lds_a, wave);
    const int ntile = blockIdx.y * 8 + wave;
    if (ntile >= NT1) return;                          // wave-uniform; EXEC stays full
    v8f z = {0.f,0.f,0.f,0.f,0.f,0.f,0.f,0.f};
    v8f acc[4] = { z, z, z, z };
    wmma_64x16(lds_a, Bf1 + (size_t)ntile * 2048, lane, acc);
    const int n  = ntile * 16 + (lane & 15);
    const float bn = bias1[n];
    #pragma unroll
    for (int ms = 0; ms < 4; ++ms) {
        const int m0 = mt * MROWS + ms * 16 + (lane >> 4) * 8;
        #pragma unroll
        for (int r = 0; r < 8; ++r)
            xp_om[(size_t)(m0 + r) * N1 + n] = acc[ms][r] + bn;
    }
}

// GEMM 2: output projection, D scattered to [B][C][HW] (bohw) + bias.
__global__ void __launch_bounds__(256)
gemm_out(const unsigned short* __restrict__ Aval,
         const unsigned* __restrict__ Bf2,
         const float* __restrict__ b_out,
         float* __restrict__ out)
{
    __shared__ unsigned short lds_a[MROWS * C_IN];     // 32 KB
    const int tid = threadIdx.x, mt = blockIdx.x;
    const int wave = tid >> 5, lane = tid & 31;
    stage_a_tdm(Aval, mt, lds_a, wave);
    const int ntile = blockIdx.y * 8 + wave;           // 0..15, all valid
    v8f z = {0.f,0.f,0.f,0.f,0.f,0.f,0.f,0.f};
    v8f acc[4] = { z, z, z, z };
    wmma_64x16(lds_a, Bf2 + (size_t)ntile * 2048, lane, acc);
    const int n  = ntile * 16 + (lane & 15);
    const float bn = b_out[n];
    const int bidx = mt / (HW / MROWS);                // 2304 % 64 == 0
    const int hwb  = (mt % (HW / MROWS)) * MROWS;
    float* outn = out + ((size_t)bidx * C_IN + n) * HW;
    #pragma unroll
    for (int ms = 0; ms < 4; ++ms) {
        const int hw0 = hwb + ms * 16 + (lane >> 4) * 8;
        #pragma unroll
        for (int r = 0; r < 8; ++r)
            outn[hw0 + r] = acc[ms][r] + bn;
    }
}

// ---------------------------------------------------------------------------
// Deformable bilinear sampling: one thread per (pixel, group).
// ---------------------------------------------------------------------------
__global__ void __launch_bounds__(256)
deform_sample(const float* __restrict__ xp_om, unsigned short* __restrict__ val)
{
    const int t  = blockIdx.x * 256 + threadIdx.x;   // m*16 + g
    const int g  = t & (GRP - 1);
    const int m  = t >> 4;
    const int b  = m / HW;
    const int hw = m % HW;
    const int h  = hw / W_DIM;
    const int w  = hw % W_DIM;

    const float* om  = xp_om + (size_t)m * N1 + C_IN + g * 27;
    const float* xpg = xp_om + (size_t)b * HW * N1 + g * GC;   // row stride N1

    float acc[GC];
    #pragma unroll
    for (int c = 0; c < GC; ++c) acc[c] = 0.f;

    #pragma unroll
    for (int k = 0; k < 9; ++k) {
        const float offx = om[2 * k];
        const float offy = om[2 * k + 1];
        const float mk   = om[18 + k];
        const float sy = (float)(h + k / 3) + offy;   // padded coords
        const float sx = (float)(w + k % 3) + offx;
        const float y0f = floorf(sy), x0f = floorf(sx);
        const int   y0 = (int)y0f,   x0 = (int)x0f;
        const float wy1 = sy - y0f, wx1 = sx - x0f;
        const float wy0 = 1.f - wy1, wx0 = 1.f - wx1;
        const float tw[4] = { wy0 * wx0, wy0 * wx1, wy1 * wx0, wy1 * wx1 };
        const int   ys[4] = { y0, y0, y0 + 1, y0 + 1 };
        const int   xs[4] = { x0, x0 + 1, x0, x0 + 1 };
        #pragma unroll
        for (int tap = 0; tap < 4; ++tap) {
            const int yi = ys[tap], xi = xs[tap];
            // padded image is zero outside 1..48 in both dims
            if (yi >= 1 && yi <= H_DIM && xi >= 1 && xi <= W_DIM) {
                const float wt = tw[tap] * mk;
                const float4* s4 =
                    (const float4*)(xpg + (size_t)((yi - 1) * W_DIM + (xi - 1)) * N1);
                #pragma unroll
                for (int q = 0; q < 4; ++q) {
                    const float4 v = s4[q];
                    acc[q * 4 + 0] += wt * v.x;
                    acc[q * 4 + 1] += wt * v.y;
                    acc[q * 4 + 2] += wt * v.z;
                    acc[q * 4 + 3] += wt * v.w;
                }
            }
        }
    }
    unsigned short* dst = val + (size_t)m * C_IN + g * GC;
    #pragma unroll
    for (int c = 0; c < GC; ++c) dst[c] = f32_to_bf16(acc[c]);
}

// ---------------------------------------------------------------------------
extern "C" void kernel_launch(void* const* d_in, const int* in_sizes, int n_in,
                              void* d_out, int out_size, void* d_ws, size_t ws_size,
                              hipStream_t stream)
{
    (void)in_sizes; (void)n_in; (void)out_size; (void)ws_size;
    const float* x       = (const float*)d_in[0];
    const float* w_value = (const float*)d_in[1];
    const float* b_value = (const float*)d_in[2];
    const float* w_om    = (const float*)d_in[3];
    const float* b_om    = (const float*)d_in[4];
    const float* w_out   = (const float*)d_in[5];
    const float* b_out   = (const float*)d_in[6];

    char* ws = (char*)d_ws;
    size_t off = 0;
    auto take = [&](size_t bytes) {
        char* p = ws + off;
        off = (off + bytes + 255) & ~(size_t)255;
        return p;
    };
    unsigned short* A1    = (unsigned short*)take((size_t)M_TOTAL * C_IN * 2); // 9.4 MB
    unsigned*       Bf1   = (unsigned*)take((size_t)NT1 * 2048 * 4);           // 352 KB
    unsigned*       Bf2   = (unsigned*)take((size_t)NT2 * 2048 * 4);           // 128 KB
    float*          bias1 = (float*)take((size_t)N1 * 4);
    float*          xp_om = (float*)take((size_t)M_TOTAL * N1 * 4);            // 50.7 MB
    unsigned short* valb  = (unsigned short*)take((size_t)M_TOTAL * C_IN * 2); // 9.4 MB

    transpose_pack<<<dim3(HW / 32, C_IN / 32, BATCH), dim3(32, 8), 0, stream>>>(x, A1);

    const int pack_total = NT1 * 2048 + NT2 * 2048 + N1;
    pack_weights<<<(pack_total + 255) / 256, 256, 0, stream>>>(
        w_value, b_value, w_om, b_om, w_out, Bf1, Bf2, bias1);

    gemm_xp_om<<<dim3(M_TOTAL / MROWS, (NT1 + 7) / 8), 256, 0, stream>>>(
        A1, Bf1, bias1, xp_om);

    deform_sample<<<(M_TOTAL * GRP) / 256, 256, 0, stream>>>(xp_om, valb);

    gemm_out<<<dim3(M_TOTAL / MROWS, NT2 / 8), 256, 0, stream>>>(
        valb, Bf2, b_out, (float*)d_out);
}
